// WeightedCorrelationBlock_10488310136850
// MI455X (gfx1250) — matrix operand order repaired and checked by reading, not compiled, not run on hardware
//
#include <hip/hip_runtime.h>

// ---------------------------------------------------------------------------
// WeightedCorrelationBlock for MI455X (gfx1250, wave32, WMMA bf16, TDM)
//   K1: encoder 1x1 conv (GEMM 64x256xN) via v_wmma_f32_16x16x32_bf16 + BN/ReLU
//   K2: fused correlation (49 offsets, group-mean) -> LDS -> decoder GEMM
//       (256x224xN) via WMMA + BN + residual + ReLU. x_corr never hits HBM.
//   Weight staging global->LDS uses the Tensor Data Mover when available.
// ---------------------------------------------------------------------------

typedef __attribute__((ext_vector_type(16))) __bf16 v16bf;
typedef __attribute__((ext_vector_type(8)))  __bf16 v8bf;
typedef __attribute__((ext_vector_type(8)))  float  v8f;
typedef unsigned int u32x4 __attribute__((ext_vector_type(4)));
typedef int          i32x8 __attribute__((ext_vector_type(8)));
typedef int          i32x4 __attribute__((ext_vector_type(4)));

#define EPSV 1e-5f
#define HW   3136      // 56*56
#define THW  50176     // 16*3136

#if defined(__has_builtin)
#if __has_builtin(__builtin_amdgcn_tensor_load_to_lds)
#define HAVE_TDM 1
#endif
#endif

__device__ __forceinline__ unsigned short f2bf(float f) {
    unsigned u = __float_as_uint(f);
    u += 0x7FFFu + ((u >> 16) & 1u);      // round-to-nearest-even
    return (unsigned short)(u >> 16);
}
__device__ __forceinline__ float bf2f(unsigned short h) {
    return __uint_as_float(((unsigned)h) << 16);
}
__device__ __forceinline__ v16bf frag_join(v8bf lo, v8bf hi) {
    return __builtin_shufflevector(lo, hi, 0,1,2,3,4,5,6,7,8,9,10,11,12,13,14,15);
}
// A fragment (16x32 bf16, row-major LDS [m][k]); kb includes +8 for lanes>=16.
__device__ __forceinline__ v16bf load_fragA(const unsigned short* row, int kb) {
    v8bf lo = *(const v8bf*)(row + kb);
    v8bf hi = *(const v8bf*)(row + kb + 16);
    return frag_join(lo, hi);
}
// B fragment (32x16 bf16, n-major LDS [n][k]); kb includes +8 for lanes>=16.
__device__ __forceinline__ v16bf load_fragB(const unsigned short* row, int kb) {
    v8bf lo = *(const v8bf*)(row + kb);
    v8bf hi = *(const v8bf*)(row + kb + 16);
    return frag_join(lo, hi);
}

// ---------------------------------------------------------------------------
// Tensor Data Mover: 1-D contiguous copy of `nelem` 2-byte elements to LDS.
// D# per CDNA5 ISA ch.8: group0 {count=1, lds_addr, global_addr, type=2},
// group1 {data_size=2B, tensor_dim0=stride0=nelem, tile=(nelem,1)}.
// EXEC is ignored for TDM; issue from one wave, complete w/ s_wait_tensorcnt.
// ---------------------------------------------------------------------------
#ifdef HAVE_TDM
__device__ __forceinline__ void tdm_load_1d(void* lds_dst, const void* gsrc,
                                            unsigned nelem) {
    unsigned lds = (unsigned)(size_t)lds_dst;            // low 32b = LDS offset
    unsigned long long ga = (unsigned long long)(size_t)gsrc;
    u32x4 g0;
    g0[0] = 1u;                                           // count=1 (valid)
    g0[1] = lds;                                          // lds_addr
    g0[2] = (unsigned)(ga & 0xFFFFFFFFu);                 // global_addr[31:0]
    g0[3] = (unsigned)((ga >> 32) & 0x01FFFFFFu) | (2u << 30); // addr[56:32]|type=2
    i32x8 g1;
    g1[0] = (int)(1u << 16);                              // data_size=1 -> 2B
    g1[1] = (int)((nelem & 0xFFFFu) << 16);               // tensor_dim0[15:0]
    g1[2] = (int)(((nelem >> 16) & 0xFFFFu) | (1u << 16)); // td0 hi | tensor_dim1=1
    g1[3] = (int)((nelem & 0xFFFFu) << 16);               // tile_dim0 = nelem
    g1[4] = 1;                                            // tile_dim1 = 1
    g1[5] = (int)nelem;                                   // tensor_dim0_stride lo
    g1[6] = 0;                                            // stride hi | td1_stride lo
    g1[7] = 0;
    i32x4 z4 = {0, 0, 0, 0};
#if __clang_major__ >= 23
    i32x8 z8 = {0, 0, 0, 0, 0, 0, 0, 0};
    __builtin_amdgcn_tensor_load_to_lds(g0, g1, z4, z4, z8, 0);
#else
    __builtin_amdgcn_tensor_load_to_lds(g0, g1, z4, z4, 0);
#endif
}
__device__ __forceinline__ void tdm_wait() {
#if __has_builtin(__builtin_amdgcn_s_wait_tensorcnt)
    __builtin_amdgcn_s_wait_tensorcnt(0);
#else
    asm volatile("s_wait_tensorcnt 0x0" ::: "memory");
#endif
}
#endif

// ---------------------------------------------------------------------------
// Kernel 0: weight conversion + BN folding
// ---------------------------------------------------------------------------
__global__ void wcb_prep_kernel(const float* __restrict__ enc_w,
                                const float* __restrict__ eg, const float* __restrict__ eb,
                                const float* __restrict__ em, const float* __restrict__ ev,
                                const float* __restrict__ dec_w,
                                const float* __restrict__ dg, const float* __restrict__ db,
                                const float* __restrict__ dm, const float* __restrict__ dv,
                                unsigned short* __restrict__ encw_bf,
                                unsigned short* __restrict__ decw_bf,
                                float* __restrict__ einv, float* __restrict__ ebias,
                                float* __restrict__ dinv, float* __restrict__ dbias) {
    const int tid = threadIdx.x;
    const int blk = blockIdx.x;
    if (blk < 256) {
        for (int k = tid; k < 224; k += 256) {            // K2G 196 -> pad 224
            float w = (k < 196) ? dec_w[blk * 196 + k] : 0.f;
            decw_bf[blk * 224 + k] = f2bf(w);
        }
    } else {
        for (int i = tid; i < 64 * 256; i += 256) encw_bf[i] = f2bf(enc_w[i]);
        if (tid < 64) {
            float iv = eg[tid] * rsqrtf(ev[tid] + EPSV);
            einv[tid]  = iv;
            ebias[tid] = eb[tid] - em[tid] * iv;
        }
        float iv = dg[tid] * rsqrtf(dv[tid] + EPSV);
        dinv[tid]  = iv;
        dbias[tid] = db[tid] - dm[tid] * iv;
    }
}

// ---------------------------------------------------------------------------
// Kernel 1: encoder  x_enc = relu(bn(enc_w @ x))  -> bf16 in ws
//   grid (784 n-tiles, 4 batches), 128 threads = 4 waves
// ---------------------------------------------------------------------------
__global__ void __launch_bounds__(128)
wcb_encoder_kernel(const float* __restrict__ x,
                   const unsigned short* __restrict__ encw_bf,
                   const float* __restrict__ einv, const float* __restrict__ ebias,
                   unsigned short* __restrict__ xe) {
    extern __shared__ char smem[];
    unsigned short* sA   = (unsigned short*)smem;    // 64 x 256 bf16 weights
    unsigned short* sB   = sA + 64 * 256;            // 64 x 256 bf16 x-tile (n-major)
    float*          sInv = (float*)(sB + 64 * 256);  // 64
    float*          sBia = sInv + 64;                // 64

    const int tid  = threadIdx.x;
    const int lane = tid & 31;
    const int wv   = tid >> 5;
    const int n0   = blockIdx.x * 64;
    const int b    = blockIdx.y;

#ifdef HAVE_TDM
    if (wv == 0) tdm_load_1d(sA, encw_bf, 64 * 256);      // 32 KB via TDM
#else
    {
        const uint4* src = (const uint4*)encw_bf;
        uint4*       dst = (uint4*)sA;
        #pragma unroll
        for (int i = 0; i < 16; ++i) dst[i * 128 + tid] = src[i * 128 + tid];
    }
#endif
    if (tid < 64) { sInv[tid] = einv[tid]; sBia[tid] = ebias[tid]; }

    // stage x tile: global [c][n] f32 (coalesced over n) -> LDS [n][c] bf16
    {
        const int n  = tid & 63;
        const int ch = (tid >> 6) * 4;
        const float* bx = x + ((long)b * 256) * THW + n0 + n;
        for (int c0 = 0; c0 < 256; c0 += 8) {
            const int c = c0 + ch;
            if (c0 + 8 < 256)
                __builtin_prefetch(bx + (long)(c + 8) * THW, 0, 1);
            float v0 = bx[(long)(c + 0) * THW];
            float v1 = bx[(long)(c + 1) * THW];
            float v2 = bx[(long)(c + 2) * THW];
            float v3 = bx[(long)(c + 3) * THW];
            unsigned p0 = (unsigned)f2bf(v0) | ((unsigned)f2bf(v1) << 16);
            unsigned p1 = (unsigned)f2bf(v2) | ((unsigned)f2bf(v3) << 16);
            *(uint2*)(sB + n * 256 + c) = make_uint2(p0, p1);
        }
    }
#ifdef HAVE_TDM
    if (wv == 0) tdm_wait();
#endif
    __syncthreads();

    const int kaA = (lane & 16) ? 8 : 0;
    const int kaB = (lane & 16) ? 8 : 0;
    const int nlo = lane & 15;
    const unsigned short* arow = sA + (wv * 16 + nlo) * 256;

    // ks-outer / nt-inner: 4 independent accumulator chains, A loaded once/ks
    const v8f vzero = {0.f, 0.f, 0.f, 0.f, 0.f, 0.f, 0.f, 0.f};
    v8f acc[4];
    #pragma unroll
    for (int i = 0; i < 4; ++i) acc[i] = vzero;
    #pragma unroll
    for (int ks = 0; ks < 8; ++ks) {
        v16bf a = load_fragA(arow, ks * 32 + kaA);
        #pragma unroll
        for (int nt = 0; nt < 4; ++nt) {
            v16bf bm = load_fragB(sB + (nt * 16 + nlo) * 256, ks * 32 + kaB);
            acc[nt] = __builtin_amdgcn_wmma_f32_16x16x32_bf16(
                          false, a, false, bm, (short)0, acc[nt], false, false);
        }
    }
    const int mo = (lane & 16) ? 8 : 0;
    #pragma unroll
    for (int nt = 0; nt < 4; ++nt) {
        #pragma unroll
        for (int r = 0; r < 8; ++r) {
            const int m = wv * 16 + mo + r;
            float v = acc[nt][r] * sInv[m] + sBia[m];
            v = v > 0.f ? v : 0.f;
            xe[((long)b * 64 + m) * THW + n0 + (nt * 16 + nlo)] = f2bf(v);
        }
    }
}

// ---------------------------------------------------------------------------
// Kernel 2: fused correlation + decoder GEMM + BN + residual + ReLU
//   grid (49 spatial tiles, 16 frames, 4 batches), 256 threads = 8 waves
//   Channel-fastest LDS layouts (row pitch 68 floats) for float4 reduction.
// ---------------------------------------------------------------------------
#define PITCH 68
__global__ void __launch_bounds__(256)
wcb_corr_decoder_kernel(const float* __restrict__ x,
                        const unsigned short* __restrict__ xe,
                        const float* __restrict__ fw,          // (64,16,7,7)
                        const unsigned short* __restrict__ decw_bf,
                        const float* __restrict__ dinv, const float* __restrict__ dbias,
                        float* __restrict__ out) {
    extern __shared__ char smem[];
    unsigned short* sW   = (unsigned short*)smem;     // 256 x 224 dec weights
    unsigned short* sB   = sW + 256 * 224;            // 64 x 224 corr tile (n-major)
    float*          sX2  = (float*)(sB + 64 * 224);   // 196 x PITCH (halo, c-fastest)
    float*          sX1  = sX2 + 196 * PITCH;         //  64 x PITCH (prev frame)
    float*          sWt  = sX1 + 64 * PITCH;          //  49 x PITCH (filter @ t)
    float*          sInv = sWt + 49 * PITCH;          // 256
    float*          sBia = sInv + 256;                // 256

    const int tid  = threadIdx.x;
    const int lane = tid & 31;
    const int wv   = tid >> 5;
    const int ty = blockIdx.x / 7, tx = blockIdx.x % 7;
    const int t  = blockIdx.y, b = blockIdx.z;
    const int h0 = ty * 8, w0 = tx * 8;

#ifdef HAVE_TDM
    if (wv == 0) tdm_load_1d(sW, decw_bf, 256 * 224); // 112 KB via TDM
#else
    {
        const uint4* src = (const uint4*)decw_bf;
        uint4*       dst = (uint4*)sW;
        #pragma unroll
        for (int i = 0; i < 28; ++i) dst[i * 256 + tid] = src[i * 256 + tid];
    }
#endif
    sInv[tid] = dinv[tid];
    sBia[tid] = dbias[tid];

    // x2: current frame, 14x14 halo tile (border zero-padded) -> [s][c]
    {
        const unsigned short* xeb = xe + (((long)b * 64) * 16 + t) * HW;
        for (int idx = tid; idx < 64 * 196; idx += 256) {
            const int c  = idx / 196, s = idx - c * 196;
            const int hy = s / 14,    hx = s - hy * 14;
            const int h = h0 + hy - 3, w = w0 + hx - 3;
            float v = 0.f;
            if (h >= 0 && h < 56 && w >= 0 && w < 56)
                v = bf2f(xeb[(long)c * THW + h * 56 + w]);
            sX2[s * PITCH + c] = v;
        }
    }
    // x1: previous frame (t clamped), center 8x8 -> [pos][c]
    {
        const int t1 = (t > 0) ? t - 1 : 0;
        const unsigned short* xeb = xe + (((long)b * 64) * 16 + t1) * HW;
        for (int idx = tid; idx < 64 * 64; idx += 256) {
            const int c = idx >> 6, pos = idx & 63;
            const int h = h0 + (pos >> 3), w = w0 + (pos & 7);
            sX1[pos * PITCH + c] = bf2f(xeb[(long)c * THW + h * 56 + w]);
        }
    }
    // filter weights for this frame: fw[c][t][o] -> [o][c]
    for (int idx = tid; idx < 64 * 49; idx += 256) {
        const int c = idx / 49, o = idx - c * 49;
        sWt[o * PITCH + c] = fw[c * 784 + t * 49 + o];
    }
#ifdef HAVE_TDM
    if (wv == 0) tdm_wait();
#endif
    __syncthreads();

    // correlation: thread = (pos, group); 16-ch reduce via float4 LDS loads
    {
        const int pos = tid >> 2, g = tid & 3;
        const int py = pos >> 3, px = pos & 7;
        const int c0 = g * 16;
        // x1 slice is offset-invariant: hoist 16 channels into registers
        float4 x1v[4];
        #pragma unroll
        for (int q = 0; q < 4; ++q)
            x1v[q] = *(const float4*)(sX1 + pos * PITCH + c0 + 4 * q);
        unsigned short* brow = sB + pos * 224;
        for (int ky = 0; ky < 7; ++ky) {
            for (int kx = 0; kx < 7; ++kx) {
                const int o = ky * 7 + kx;
                const float* wp = sWt + o * PITCH + c0;
                const float* a2 = sX2 + ((py + ky) * 14 + (px + kx)) * PITCH + c0;
                float acc = 0.f;
                #pragma unroll
                for (int q = 0; q < 4; ++q) {
                    float4 wv4 = *(const float4*)(wp + 4 * q);
                    float4 a2v = *(const float4*)(a2 + 4 * q);
                    acc += wv4.x * x1v[q].x * a2v.x;
                    acc += wv4.y * x1v[q].y * a2v.y;
                    acc += wv4.z * x1v[q].z * a2v.z;
                    acc += wv4.w * x1v[q].w * a2v.w;
                }
                brow[o * 4 + g] = f2bf(acc * 0.0625f);   // mean over 16 ch
            }
        }
        #pragma unroll
        for (int i = 0; i < 7; ++i) brow[196 + g + i * 4] = 0;  // K pad
    }
    __syncthreads();

    // decoder GEMM: wave wv -> rows [wv*32, wv*32+32); 8 independent acc chains
    const int m0  = wv * 32;
    const int kaA = (lane & 16) ? 8 : 0;
    const int kaB = (lane & 16) ? 8 : 0;
    const int nlo = lane & 15;
    const unsigned short* arow0 = sW + (m0 + nlo) * 224;
    const unsigned short* arow1 = arow0 + 16 * 224;
    const unsigned short* brow0 = sB + nlo * 224;

    const v8f vzero = {0.f, 0.f, 0.f, 0.f, 0.f, 0.f, 0.f, 0.f};
    v8f acc[8];
    #pragma unroll
    for (int i = 0; i < 8; ++i) acc[i] = vzero;

    #pragma unroll
    for (int ks = 0; ks < 7; ++ks) {
        const int ka = ks * 32;
        v16bf a0 = load_fragA(arow0, ka + kaA);
        v16bf a1 = load_fragA(arow1, ka + kaA);
        #pragma unroll
        for (int nt = 0; nt < 4; ++nt) {
            v16bf bm = load_fragB(brow0 + nt * 16 * 224, ka + kaB);
            acc[nt]     = __builtin_amdgcn_wmma_f32_16x16x32_bf16(
                              false, a0, false, bm, (short)0, acc[nt], false, false);
            acc[4 + nt] = __builtin_amdgcn_wmma_f32_16x16x32_bf16(
                              false, a1, false, bm, (short)0, acc[4 + nt], false, false);
        }
    }

    // epilogue: BN + residual + ReLU
    const int mo = (lane & 16) ? 8 : 0;
    #pragma unroll
    for (int mt = 0; mt < 2; ++mt) {
        #pragma unroll
        for (int nt = 0; nt < 4; ++nt) {
            #pragma unroll
            for (int r = 0; r < 8; ++r) {
                const int m   = m0 + mt * 16 + mo + r;
                const int pos = nt * 16 + nlo;
                const int h = h0 + (pos >> 3), w = w0 + (pos & 7);
                const long gi = ((long)b * 256 + m) * THW + t * HW + h * 56 + w;
                float v = acc[mt * 4 + nt][r] * sInv[m] + sBia[m] + x[gi];
                out[gi] = v > 0.f ? v : 0.f;
            }
        }
    }
}

// ---------------------------------------------------------------------------
// Workspace layout (bytes):
//   [0,32768)            enc_w bf16   (64x256)
//   [32768,147456)       dec_w bf16   (256x224, K-padded)
//   [147456..150016)     folded BN tables
//   [150272, +25690112)  x_enc bf16   (4x64x16x3136)      total ~25.9 MB
// ---------------------------------------------------------------------------
extern "C" void kernel_launch(void* const* d_in, const int* in_sizes, int n_in,
                              void* d_out, int out_size, void* d_ws, size_t ws_size,
                              hipStream_t stream) {
    const float* x   = (const float*)d_in[0];
    const float* ew  = (const float*)d_in[1];
    const float* eg  = (const float*)d_in[2];
    const float* eb  = (const float*)d_in[3];
    const float* em  = (const float*)d_in[4];
    const float* ev  = (const float*)d_in[5];
    const float* fw  = (const float*)d_in[6];
    const float* dw  = (const float*)d_in[7];
    const float* dg  = (const float*)d_in[8];
    const float* db  = (const float*)d_in[9];
    const float* dm  = (const float*)d_in[10];
    const float* dv  = (const float*)d_in[11];
    float* out = (float*)d_out;

    char* ws = (char*)d_ws;
    unsigned short* encw_bf = (unsigned short*)(ws + 0);
    unsigned short* decw_bf = (unsigned short*)(ws + 32768);
    float* einv  = (float*)(ws + 147456);
    float* ebias = (float*)(ws + 147712);
    float* dinv  = (float*)(ws + 147968);
    float* dbias = (float*)(ws + 148992);
    unsigned short* xe = (unsigned short*)(ws + 150272);

    wcb_prep_kernel<<<257, 256, 0, stream>>>(ew, eg, eb, em, ev,
                                             dw, dg, db, dm, dv,
                                             encw_bf, decw_bf,
                                             einv, ebias, dinv, dbias);

    const size_t smem1 = 64 * 256 * 2 * 2 + 64 * 2 * 4;               // 66,048 B
    wcb_encoder_kernel<<<dim3(784, 4), 128, smem1, stream>>>(x, encw_bf, einv, ebias, xe);

    const size_t smem2 = 256 * 224 * 2 + 64 * 224 * 2 +
                         (196 + 64 + 49) * PITCH * 4 + 256 * 2 * 4;   // 229,456 B
    wcb_corr_decoder_kernel<<<dim3(49, 16, 4), 256, smem2, stream>>>(
        x, xe, fw, decw_bf, dinv, dbias, out);
}
